// RNN_83476984365506
// MI455X (gfx1250) — matrix operand order repaired and checked by reading, not compiled, not run on hardware
//
#include <hip/hip_runtime.h>

// Problem constants (match reference)
#define BB 256
#define NN 512
#define NI 4
#define TT 1000

#define P_OLD 0.8f
#define P_NEW 0.2f
#define NOISE_CONST 0.063245553203367586f  // sqrt(2*0.01/5)

typedef __attribute__((ext_vector_type(16))) __bf16 v16bf;
typedef __attribute__((ext_vector_type(8)))  float  v8f;

// LDS row stride for staged A (relu(act) in bf16), padded to avoid bank conflicts
#define ASTRIDE 528   // 512 + 16 elems -> 1056B row stride

__global__ __launch_bounds__(256) void rnn_zero_kernel(float* __restrict__ p) {
    p[blockIdx.x * 256 + threadIdx.x] = 0.0f;
}

__global__ __launch_bounds__(256) void rnn_prep_wrec(const float* __restrict__ Wrec,
                                                     __bf16* __restrict__ Wb) {
    int i = blockIdx.x * 256 + threadIdx.x;
    Wb[i] = (__bf16)Wrec[i];
}

// One recurrence step:
//   act_out[b,i] = P_OLD*act_in[b,i]
//                + P_NEW*( sum_j relu(act_in[b,j])*Wrec[i,j] + sum_q inputs[b,q,t]*inp_w[q,i] )
//                + NOISE_CONST*noise_t[b,i]
// grid = 64 blocks x 256 threads (8 waves). Block -> m_tile = blockIdx&15 (batch rows),
// wave w -> n_tile = (blockIdx>>4)*8 + w (neuron cols). One 16x16 tile per wave.
__global__ __launch_bounds__(256)
void rnn_step_kernel(const float* __restrict__ act_in,
                     float*       __restrict__ act_out,
                     const __bf16* __restrict__ Wb,       // Wrec in bf16, row-major [N][N]
                     const float* __restrict__ noise_t,   // [B][N] slice for this t
                     const float* __restrict__ inputs,    // [B][NI][T]
                     int t,
                     const float* __restrict__ inp_w)     // [NI][N]
{
    __shared__ __bf16 As[16 * ASTRIDE];

    const int lane   = threadIdx.x & 31;
    const int wave   = threadIdx.x >> 5;          // 0..7
    const int m_tile = blockIdx.x & 15;           // 16 batch tiles
    const int n_tile = ((blockIdx.x >> 4) << 3) + wave;  // 0..31

    // Stage relu(act) rows for this block's batch tile into LDS as bf16.
    // 16 rows x 512 cols, contiguous fp32 in global.
    {
        const float* actRows = act_in + (size_t)m_tile * 16 * NN;
        for (int idx = threadIdx.x; idx < 16 * NN; idx += 256) {
            int r = idx >> 9;          // /512
            int c = idx & (NN - 1);
            float v = actRows[idx];
            As[r * ASTRIDE + c] = (__bf16)(v > 0.0f ? v : 0.0f);
        }
    }
    __syncthreads();

    const int rt = lane & 15;      // row-in-tile (A: M, B: N-col)
    const int hi = lane >> 4;      // lane-half selector

    v8f acc = {};
    for (int k = 0; k < NN; k += 32) {
        // ---- A fragment (16-bit A 16x32 layout):
        // lanes 0-15:  elems 0..7 -> K=k+0..7,   elems 8..15 -> K=k+16..23
        // lanes 16-31: elems 0..7 -> K=k+8..15,  elems 8..15 -> K=k+24..31
        v16bf a;
        {
            const int abase = rt * ASTRIDE + k + hi * 8;
#pragma unroll
            for (int e = 0; e < 8; ++e) {
                a[e]     = As[abase + e];
                a[e + 8] = As[abase + 16 + e];
            }
        }
        // ---- B fragment (16-bit B 32x16 layout): B[k,n] = Wrec[n,k]
        // lanes 0-15: elems 0..7 -> K=k+0..7, elems 8..15 -> K=k+16..23 ; lanes 16-31: K=k+8..15, K=k+24..31 ; n = rt
        v16bf b;
        {
            const __bf16* wrow = Wb + (size_t)(n_tile * 16 + rt) * NN + k + hi * 8;
#pragma unroll
            for (int e = 0; e < 8; ++e) { b[e] = wrow[e]; b[e + 8] = wrow[16 + e]; }
        }
        acc = __builtin_amdgcn_wmma_f32_16x16x32_bf16(
            /*neg_a=*/false, a, /*neg_b=*/false, b,
            /*c_mod=*/(short)0, acc, /*reuse_a=*/false, /*reuse_b=*/false);
    }

    // Epilogue. C/D layout: VGPR r, lanes 0-15 -> M=r, lanes 16-31 -> M=r+8; N=lane&15.
    const int ncol = n_tile * 16 + rt;
    const float w0 = inp_w[0 * NN + ncol];
    const float w1 = inp_w[1 * NN + ncol];
    const float w2 = inp_w[2 * NN + ncol];
    const float w3 = inp_w[3 * NN + ncol];
#pragma unroll
    for (int r = 0; r < 8; ++r) {
        const int mrow = m_tile * 16 + (hi ? r + 8 : r);
        const float* xb = inputs + (size_t)mrow * (NI * TT) + t;
        const float xin = xb[0] * w0 + xb[TT] * w1 + xb[2 * TT] * w2 + xb[3 * TT] * w3;
        const size_t off = (size_t)mrow * NN + ncol;
        const float a_old = act_in[off];
        const float nz    = noise_t[off];
        act_out[off] = P_OLD * a_old + P_NEW * (acc[r] + xin) + NOISE_CONST * nz;
    }
}

// out[b] = sum_i act[b,i]*out_w[i]; also copy act into d_out[B ..].
__global__ __launch_bounds__(128)
void rnn_finalize_kernel(const float* __restrict__ act,
                         const float* __restrict__ out_w,
                         float* __restrict__ out) {
    __shared__ float red[128];
    const int b = blockIdx.x;
    float s = 0.0f;
    for (int i = threadIdx.x; i < NN; i += 128) {
        float v = act[(size_t)b * NN + i];
        s += v * out_w[i];
        out[BB + (size_t)b * NN + i] = v;
    }
    red[threadIdx.x] = s;
    __syncthreads();
    for (int off = 64; off > 0; off >>= 1) {
        if (threadIdx.x < off) red[threadIdx.x] += red[threadIdx.x + off];
        __syncthreads();
    }
    if (threadIdx.x == 0) out[b] = red[0];
}

extern "C" void kernel_launch(void* const* d_in, const int* in_sizes, int n_in,
                              void* d_out, int out_size, void* d_ws, size_t ws_size,
                              hipStream_t stream) {
    const float* inputs = (const float*)d_in[0];  // [B, NI, T]
    const float* noise  = (const float*)d_in[1];  // [T, B, N]
    const float* inp_w  = (const float*)d_in[2];  // [NI, N]
    const float* Wrec   = (const float*)d_in[3];  // [N, N]
    const float* out_w  = (const float*)d_in[4];  // [N, 1]
    float* out = (float*)d_out;                   // [B] ++ [B*N]

    // Workspace: act double buffer (fp32) + Wrec bf16
    float*  act0 = (float*)d_ws;                  // B*N
    float*  act1 = act0 + BB * NN;                // B*N
    __bf16* Wb   = (__bf16*)(act1 + BB * NN);     // N*N

    rnn_zero_kernel<<<(BB * NN) / 256, 256, 0, stream>>>(act0);
    rnn_prep_wrec<<<(NN * NN) / 256, 256, 0, stream>>>(Wrec, Wb);

    for (int t = 0; t < TT; ++t) {
        const float* ain = (t & 1) ? act1 : act0;
        float*       aout = (t & 1) ? act0 : act1;
        rnn_step_kernel<<<64, 256, 0, stream>>>(
            ain, aout, Wb, noise + (size_t)t * BB * NN, inputs, t, inp_w);
    }
    // T=1000 is even -> final act is in act0
    rnn_finalize_kernel<<<BB, 128, 0, stream>>>(act0, out_w, out);
}
